// ann1_snn2_54760833024460
// MI455X (gfx1250) — matrix operand order repaired and checked by reading, not compiled, not run on hardware
//
#include <hip/hip_runtime.h>
#include <hip/hip_bf16.h>
#include <math.h>

typedef _Float16 half_t;
typedef __attribute__((ext_vector_type(16))) _Float16 v16h;
typedef __attribute__((ext_vector_type(8)))  _Float16 v8h;
typedef __attribute__((ext_vector_type(8)))  float    v8f;

#define BB    256          // batch
#define TT    200          // time steps
#define H1    500          // hidden width
#define H1P   512          // padded hidden
#define KIN   784
#define KINP  800          // padded (multiple of 32)
#define NOUT  10
#define NOUTP 16
#define MBIG  (TT*BB)      // 51200 batched rows

static constexpr float V0C    = 2.1165344f;   // eta^(eta/(eta-1))/(eta-1), eta=4
static constexpr float SIGMAC = 0.77880078f;  // exp(-1/4)

// ---------- pad + fp32->f16 cast ----------
__global__ void pad_cast_f16(const float* __restrict__ src, half_t* __restrict__ dst,
                             int rows_src, int cols_src, int rows_dst, int cols_dst) {
  int idx = blockIdx.x * blockDim.x + threadIdx.x;
  int total = rows_dst * cols_dst;
  if (idx >= total) return;
  int r = idx / cols_dst, c = idx % cols_dst;
  float v = (r < rows_src && c < cols_src) ? src[(size_t)r * cols_src + c] : 0.0f;
  dst[idx] = (half_t)v;
}

// ---------- WMMA GEMM:  C[M,N] = A[M,K] * B[N,K]^T  (row-major, f16 in, f32 acc) ----------
// mode 0: out f32, sigmoid(acc + bias[col]) for col<nbias else 0   (ANN layer)
// mode 1: out f16, raw acc                                         (w2 = cur2 @ W2^T)
// mode 2: out f32, raw acc                                         (w3 = cur3 @ W3^T)
__global__ __launch_bounds__(256, 1) void wmma_gemm(
    const half_t* __restrict__ A, const half_t* __restrict__ Bm,
    void* __restrict__ Cout, const float* __restrict__ bias,
    int M, int N, int K, int lda, int ldb, int ldc, int mode, int nbias)
{
  const int tid  = threadIdx.x;
  const int wave = tid >> 5;
  const int lane = tid & 31;
  const int ln   = lane & 15;
  const int sel  = lane >> 4;          // 0: lanes 0-15, 1: lanes 16-31

  const int nbase = blockIdx.x * 64 + (wave & 3) * 16;
  if (nbase >= N) return;
  const int mbase = blockIdx.y * 128 + (wave >> 2) * 64;

  v8f acc[4] = {v8f{}, v8f{}, v8f{}, v8f{}};

  // B fragment: lane ln holds column (nbase+ln); K chunks [k0+sel*8 .. +7] and [k0+16+sel*8 .. +7]
  const half_t* bptr = Bm + (size_t)(nbase + ln) * ldb + sel * 8;
  // A fragments: lane ln holds row (mtile+ln); K chunks [k0+sel*8 .. +7] and [k0+16+sel*8 .. +7]
  const half_t* ap0 = A + (size_t)(mbase +  0 + ln) * lda + sel * 8;
  const half_t* ap1 = A + (size_t)(mbase + 16 + ln) * lda + sel * 8;
  const half_t* ap2 = A + (size_t)(mbase + 32 + ln) * lda + sel * 8;
  const half_t* ap3 = A + (size_t)(mbase + 48 + ln) * lda + sel * 8;

  for (int k0 = 0; k0 < K; k0 += 32) {
    v8h blo = *(const v8h*)(bptr + k0); v8h bhi = *(const v8h*)(bptr + k0 + 16);
    v16h bf = __builtin_shufflevector(blo, bhi, 0,1,2,3,4,5,6,7,8,9,10,11,12,13,14,15);

    // stream-prefetch the A operand (global_prefetch_b8); B stays L2-resident
    __builtin_prefetch(ap0 + k0 + 128, 0, 0);

    v8h lo0 = *(const v8h*)(ap0 + k0); v8h hi0 = *(const v8h*)(ap0 + k0 + 16);
    v8h lo1 = *(const v8h*)(ap1 + k0); v8h hi1 = *(const v8h*)(ap1 + k0 + 16);
    v8h lo2 = *(const v8h*)(ap2 + k0); v8h hi2 = *(const v8h*)(ap2 + k0 + 16);
    v8h lo3 = *(const v8h*)(ap3 + k0); v8h hi3 = *(const v8h*)(ap3 + k0 + 16);
    v16h af0 = __builtin_shufflevector(lo0, hi0, 0,1,2,3,4,5,6,7,8,9,10,11,12,13,14,15);
    v16h af1 = __builtin_shufflevector(lo1, hi1, 0,1,2,3,4,5,6,7,8,9,10,11,12,13,14,15);
    v16h af2 = __builtin_shufflevector(lo2, hi2, 0,1,2,3,4,5,6,7,8,9,10,11,12,13,14,15);
    v16h af3 = __builtin_shufflevector(lo3, hi3, 0,1,2,3,4,5,6,7,8,9,10,11,12,13,14,15);

    acc[0] = __builtin_amdgcn_wmma_f32_16x16x32_f16(false, af0, false, bf, (short)0, acc[0], false, false);
    acc[1] = __builtin_amdgcn_wmma_f32_16x16x32_f16(false, af1, false, bf, (short)0, acc[1], false, false);
    acc[2] = __builtin_amdgcn_wmma_f32_16x16x32_f16(false, af2, false, bf, (short)0, acc[2], false, false);
    acc[3] = __builtin_amdgcn_wmma_f32_16x16x32_f16(false, af3, false, bf, (short)0, acc[3], false, false);
  }

  const int col = nbase + ln;
#pragma unroll
  for (int mt = 0; mt < 4; ++mt) {
    int rowb = mbase + mt * 16 + sel * 8;   // C/D layout: VGPR r -> row rowb+r, lane&15 -> col
#pragma unroll
    for (int r = 0; r < 8; ++r) {
      int row = rowb + r;
      if (row >= M) continue;
      float v = acc[mt][r];
      if (mode == 0) {
        float o = 0.0f;
        if (col < nbias) o = 1.0f / (1.0f + __expf(-(v + bias[col])));
        ((float*)Cout)[(size_t)row * ldc + col] = o;
      } else if (mode == 1) {
        ((half_t*)Cout)[(size_t)row * ldc + col] = (half_t)v;
      } else {
        ((float*)Cout)[(size_t)row * ldc + col] = v;
      }
    }
  }
}

// ---------- cur2[t] dual-exp IIR, driven by constant ann; writes whole [T,B,512] f16 ----------
__global__ void cur2_iir(const float* __restrict__ ann, const float* __restrict__ a1,
                         const float* __restrict__ a2, half_t* __restrict__ cur2h) {
  int idx = blockIdx.x * blockDim.x + threadIdx.x;
  if (idx >= BB * H1P) return;
  int b = idx >> 9, k = idx & (H1P - 1);
  if (k >= H1) {
    for (int t = 0; t < TT; ++t) cur2h[(size_t)(t * BB + b) * H1P + k] = (half_t)0.0f;
    return;
  }
  float drive = V0C * ann[(size_t)b * H1P + k];
  float c1 = a1[k], c2 = a2[k];
  float p1 = 0.0f, p2 = 0.0f;
  for (int t = 0; t < TT; ++t) {
    float c = c1 * p1 + c2 * p2 + drive;
    cur2h[(size_t)(t * BB + b) * H1P + k] = (half_t)c;
    p2 = p1; p1 = c;
  }
}

// ---------- layer-2 LIF recursion + cur3 IIR, per (b,k), in-thread t loop ----------
__global__ void lif2_cur3(const half_t* __restrict__ w2h, const float* __restrict__ b2,
                          const float* __restrict__ a1, const float* __restrict__ a2,
                          half_t* __restrict__ cur3h) {
  int idx = blockIdx.x * blockDim.x + threadIdx.x;
  if (idx >= BB * H1P) return;
  int b = idx >> 9, k = idx & (H1P - 1);
  if (k >= H1) {
    for (int t = 0; t < TT; ++t) cur3h[(size_t)(t * BB + b) * H1P + k] = (half_t)0.0f;
    return;
  }
  float bk = b2[k], c1 = a1[k], c2 = a2[k];
  float v = 0.0f, s = 0.0f, p1 = 0.0f, p2 = 0.0f;
  for (int t = 0; t < TT; ++t) {
    size_t o = (size_t)(t * BB + b) * H1P + k;
    float w = (float)w2h[o] + bk;
    v = SIGMAC * v * (1.0f - s) + w;           // reset-by-spike membrane
    s = (v > 1.0f) ? 1.0f : 0.0f;              // Heaviside threshold
    float c = c1 * p1 + c2 * p2 + V0C * s;     // axon3 IIR
    cur3h[o] = (half_t)c;                      // overwrites dead cur2 buffer
    p2 = p1; p1 = c;
  }
}

// ---------- layer-3 LIF recursion; writes out[B,10,T] ----------
__global__ void lif3_out(const float* __restrict__ w3, const float* __restrict__ b3,
                         float* __restrict__ out) {
  int idx = blockIdx.x * blockDim.x + threadIdx.x;
  if (idx >= BB * NOUTP) return;
  int b = idx >> 4, j = idx & 15;
  if (j >= NOUT) return;
  float bj = b3[j];
  float v = 0.0f, s = 0.0f;
  for (int t = 0; t < TT; ++t) {
    float w = w3[(size_t)(t * BB + b) * NOUTP + j] + bj;
    v = SIGMAC * v * (1.0f - s) + w;
    s = (v > 1.0f) ? 1.0f : 0.0f;
    out[(size_t)b * NOUT * TT + (size_t)j * TT + t] = s;
  }
}

static inline int g256(long long n) { return (int)((n + 255) / 256); }

extern "C" void kernel_launch(void* const* d_in, const int* in_sizes, int n_in,
                              void* d_out, int out_size, void* d_ws, size_t ws_size,
                              hipStream_t stream) {
  (void)in_sizes; (void)n_in; (void)out_size; (void)ws_size;
  const float* inputs = (const float*)d_in[0];
  const float* W1     = (const float*)d_in[1];
  const float* b1     = (const float*)d_in[2];
  const float* a1_2   = (const float*)d_in[3];
  const float* a2_2   = (const float*)d_in[4];
  const float* W2     = (const float*)d_in[5];
  const float* b2     = (const float*)d_in[6];
  const float* a1_3   = (const float*)d_in[7];
  const float* a2_3   = (const float*)d_in[8];
  const float* W3     = (const float*)d_in[9];
  const float* b3     = (const float*)d_in[10];
  float* out = (float*)d_out;

  char* ws = (char*)d_ws;
  size_t off = 0;
  auto alloc = [&](size_t bytes) -> void* {
    void* p = ws + off;
    off = (off + bytes + 255) & ~(size_t)255;
    return p;
  };
  half_t* inputs_h = (half_t*)alloc((size_t)BB   * KINP * 2);
  half_t* W1h      = (half_t*)alloc((size_t)H1P  * KINP * 2);
  float*  ann      = (float* )alloc((size_t)BB   * H1P  * 4);
  half_t* W2h      = (half_t*)alloc((size_t)H1P  * H1P  * 2);
  half_t* W3h      = (half_t*)alloc((size_t)NOUTP* H1P  * 2);
  half_t* curh     = (half_t*)alloc((size_t)MBIG * H1P  * 2); // cur2, later reused as cur3
  half_t* w2h      = (half_t*)alloc((size_t)MBIG * H1P  * 2);
  float*  w3       = (float* )alloc((size_t)MBIG * NOUTP* 4);

  dim3 blk(256);

  // 1) pad + cast weights/inputs to f16 once
  pad_cast_f16<<<g256((long long)BB * KINP),    blk, 0, stream>>>(inputs, inputs_h, BB,  KIN, BB,    KINP);
  pad_cast_f16<<<g256((long long)H1P * KINP),   blk, 0, stream>>>(W1,     W1h,      H1,  KIN, H1P,   KINP);
  pad_cast_f16<<<g256((long long)H1P * H1P),    blk, 0, stream>>>(W2,     W2h,      H1,  H1,  H1P,   H1P);
  pad_cast_f16<<<g256((long long)NOUTP * H1P),  blk, 0, stream>>>(W3,     W3h,      NOUT,H1,  NOUTP, H1P);

  // 2) ANN front-end: ann = sigmoid(inputs @ W1^T + b1)   [256,512]
  wmma_gemm<<<dim3(H1P / 64, BB / 128), blk, 0, stream>>>(
      inputs_h, W1h, ann, b1, BB, H1P, KINP, KINP, KINP, H1P, /*mode=*/0, /*nbias=*/H1);

  // 3) precompute full cur2[t] sequence (IIR independent of spikes)
  cur2_iir<<<g256((long long)BB * H1P), blk, 0, stream>>>(ann, a1_2, a2_2, curh);

  // 4) one big batched GEMM: w2 = cur2 @ W2^T   [51200,512]
  wmma_gemm<<<dim3(H1P / 64, MBIG / 128), blk, 0, stream>>>(
      curh, W2h, w2h, nullptr, MBIG, H1P, H1P, H1P, H1P, H1P, /*mode=*/1, 0);

  // 5) LIF2 recursion + cur3 IIR (in-thread time loop), cur3 overwrites cur2 buffer
  lif2_cur3<<<g256((long long)BB * H1P), blk, 0, stream>>>(w2h, b2, a1_3, a2_3, curh);

  // 6) batched GEMM: w3 = cur3 @ W3^T   [51200,16]
  wmma_gemm<<<dim3(1, MBIG / 128), blk, 0, stream>>>(
      curh, W3h, w3, nullptr, MBIG, NOUTP, H1P, H1P, H1P, NOUTP, /*mode=*/2, 0);

  // 7) LIF3 recursion -> out[B,10,T]
  lif3_out<<<g256((long long)BB * NOUTP), blk, 0, stream>>>(w3, b3, out);
}
